// GlobalFeatureAggregation_1511828488321
// MI455X (gfx1250) — matrix operand order repaired and checked by reading, not compiled, not run on hardware
//
#include <hip/hip_runtime.h>
#include <hip/hip_bf16.h>

// ---------------------------------------------------------------------------
// GlobalFeatureAggregation for MI455X (gfx1250, wave32, WMMA bf16 16x16x32)
// B=4, C=256, H=W=64, N=4096.  Flash-attention formulation, split-bf16 QK^T.
// ---------------------------------------------------------------------------

#define B_DIM 4
#define C_DIM 256
#define N_DIM 4096
#define CR_DIM 64
#define EPS 1e-5f
#define ATTN_WAVES 4

typedef __attribute__((ext_vector_type(16))) __bf16 v16bf;
typedef __attribute__((ext_vector_type(8)))  __bf16 v8bf;
typedef __attribute__((ext_vector_type(8)))  float  v8f;

static __device__ __forceinline__ v8f wmma_bf16(v16bf a, v16bf b, v8f c) {
    // D = A(16x32 bf16) * B(32x16 bf16) + C(16x16 f32)
    return __builtin_amdgcn_wmma_f32_16x16x32_bf16(
        /*neg_a=*/false, a, /*neg_b=*/false, b,
        /*c_mod=*/(short)0, c, /*reuse_a=*/false, /*reuse_b=*/false);
}

static __device__ __forceinline__ v8f v8f_zero() {
    v8f z;
#pragma unroll
    for (int i = 0; i < 8; ++i) z[i] = 0.0f;
    return z;
}

// ---------------------------------------------------------------------------
// Kernel 1: channel LayerNorm per (b,n); write bf16 hi/lo splits.
//   xT_hi/xT_lo : (B, N, C) layout  (Q/K operand rows, contiguous in C)
//   xc_hi       : (B, C, N) layout  (V operand rows, contiguous in N)
// ---------------------------------------------------------------------------
__global__ __launch_bounds__(256) void ln_split_kernel(
    const float* __restrict__ x,
    unsigned short* __restrict__ xT_hi_u,
    unsigned short* __restrict__ xT_lo_u,
    unsigned short* __restrict__ xc_hi_u)
{
    __bf16* xT_hi = reinterpret_cast<__bf16*>(xT_hi_u);
    __bf16* xT_lo = reinterpret_cast<__bf16*>(xT_lo_u);
    __bf16* xc_hi = reinterpret_cast<__bf16*>(xc_hi_u);

    const int idx = blockIdx.x * blockDim.x + threadIdx.x;   // 0 .. B*N-1
    const int b = idx / N_DIM;
    const int n = idx % N_DIM;
    const float* xb = x + (size_t)b * C_DIM * N_DIM + n;

    float sum = 0.0f, ss = 0.0f;
    for (int c = 0; c < C_DIM; ++c) {
        float v = xb[(size_t)c * N_DIM];
        sum += v;
        ss  += v * v;
    }
    const float mu  = sum * (1.0f / C_DIM);
    const float var = ss * (1.0f / C_DIM) - mu * mu;
    const float inv = rsqrtf(var + EPS);

    __bf16* rowT_hi = xT_hi + ((size_t)b * N_DIM + n) * C_DIM;
    __bf16* rowT_lo = xT_lo + ((size_t)b * N_DIM + n) * C_DIM;
    __bf16* colc    = xc_hi + (size_t)b * C_DIM * N_DIM + n;
    for (int c = 0; c < C_DIM; ++c) {
        float v = (xb[(size_t)c * N_DIM] - mu) * inv;
        __bf16 h = (__bf16)v;
        __bf16 l = (__bf16)(v - (float)h);
        rowT_hi[c] = h;
        rowT_lo[c] = l;
        colc[(size_t)c * N_DIM] = h;
    }
}

// ---------------------------------------------------------------------------
// Kernel 2: flash attention.  One wave handles 16 query rows over all C=256
// output channels.  m processed in steps of 32 (two 16-wide S tiles).
// ---------------------------------------------------------------------------
__global__ __launch_bounds__(ATTN_WAVES * 32) void attn_flash_kernel(
    const unsigned short* __restrict__ xT_hi_u,
    const unsigned short* __restrict__ xT_lo_u,
    const unsigned short* __restrict__ xc_hi_u,
    float* __restrict__ ctx)
{
    const __bf16* xT_hi = reinterpret_cast<const __bf16*>(xT_hi_u);
    const __bf16* xT_lo = reinterpret_cast<const __bf16*>(xT_lo_u);
    const __bf16* xc_hi = reinterpret_cast<const __bf16*>(xc_hi_u);

    __shared__ __bf16 lds_p[ATTN_WAVES][16][32];   // per-wave P tile bounce

    const int lane = threadIdx.x & 31;
    const int wave = threadIdx.x >> 5;
    const int lh   = lane & 15;     // lane within half
    const int hi   = lane >> 4;     // which half of the wave

    const int wid    = blockIdx.x * ATTN_WAVES + wave;      // 0 .. B*N/16-1
    const int b      = wid / (N_DIM / 16);
    const int n_base = (wid % (N_DIM / 16)) * 16;

    const __bf16* xT_hi_b = xT_hi + (size_t)b * N_DIM * C_DIM;
    const __bf16* xT_lo_b = xT_lo + (size_t)b * N_DIM * C_DIM;
    const __bf16* xc_b    = xc_hi + (size_t)b * C_DIM * N_DIM;

    // ---- resident Q tiles (A layout), hi and lo, for the 8 K-chunks -------
    // A 16x32 bf16, this code's k order (same as B): lane holds K = 16*hi + 0..15 in elems 0..15
    // (lane<16: hi=0; lane>=16: hi=1).
    v16bf qa_hi[8], qa_lo[8];
    {
        const __bf16* rowh = xT_hi_b + (size_t)(n_base + lh) * C_DIM;
        const __bf16* rowl = xT_lo_b + (size_t)(n_base + lh) * C_DIM;
#pragma unroll
        for (int k = 0; k < 8; ++k) {
            const int base = k * 32 + 16 * hi;
            v8bf h0 = *(const v8bf*)(rowh + base);
            v8bf h1 = *(const v8bf*)(rowh + base + 8);
            v8bf l0 = *(const v8bf*)(rowl + base);
            v8bf l1 = *(const v8bf*)(rowl + base + 8);
#pragma unroll
            for (int i = 0; i < 8; ++i) {
                qa_hi[k][i] = h0[i];  qa_hi[k][8 + i] = h1[i];
                qa_lo[k][i] = l0[i];  qa_lo[k][8 + i] = l1[i];
            }
        }
    }

    // ---- output accumulators: 16 tiles (16 rows x 256 cols), C layout -----
    v8f acc[16];
#pragma unroll
    for (int t = 0; t < 16; ++t) acc[t] = v8f_zero();

    // running softmax stats, replicated across each 16-lane half:
    // reg r on lanes<16 => query row r ; on lanes>=16 => query row r+8
    float m_run[8], l_run[8];
#pragma unroll
    for (int r = 0; r < 8; ++r) { m_run[r] = -1e30f; l_run[r] = 0.0f; }

    for (int m0 = 0; m0 < N_DIM; m0 += 32) {
        // ---- S = Q * K^T for 16x32 block (two 16x16 tiles), split bf16 ----
        v8f s0 = v8f_zero(), s1 = v8f_zero();
#pragma unroll
        for (int k = 0; k < 8; ++k) {
            // B 32x16 bf16, this code's k order (same as Q): lane n holds K = 16*hi + 0..15 contiguously.
            const int coff = k * 32 + 16 * hi;
            const __bf16* kr0h = xT_hi_b + (size_t)(m0 + lh) * C_DIM + coff;
            const __bf16* kr0l = xT_lo_b + (size_t)(m0 + lh) * C_DIM + coff;
            const __bf16* kr1h = xT_hi_b + (size_t)(m0 + 16 + lh) * C_DIM + coff;
            const __bf16* kr1l = xT_lo_b + (size_t)(m0 + 16 + lh) * C_DIM + coff;
            v16bf b0h = *(const v16bf*)kr0h;
            v16bf b0l = *(const v16bf*)kr0l;
            v16bf b1h = *(const v16bf*)kr1h;
            v16bf b1l = *(const v16bf*)kr1l;
            s0 = wmma_bf16(qa_hi[k], b0h, s0);
            s0 = wmma_bf16(qa_hi[k], b0l, s0);
            s0 = wmma_bf16(qa_lo[k], b0h, s0);
            s1 = wmma_bf16(qa_hi[k], b1h, s1);
            s1 = wmma_bf16(qa_hi[k], b1l, s1);
            s1 = wmma_bf16(qa_lo[k], b1h, s1);
        }

        // ---- online softmax in native C layout --------------------------
#pragma unroll
        for (int r = 0; r < 8; ++r) {
            float mx = fmaxf(s0[r], s1[r]);
            mx = fmaxf(mx, __shfl_xor(mx, 1, 32));
            mx = fmaxf(mx, __shfl_xor(mx, 2, 32));
            mx = fmaxf(mx, __shfl_xor(mx, 4, 32));
            mx = fmaxf(mx, __shfl_xor(mx, 8, 32));
            const float mnew  = fmaxf(m_run[r], mx);
            const float scale = __expf(m_run[r] - mnew);
            m_run[r] = mnew;
            const float p0 = __expf(s0[r] - mnew);
            const float p1 = __expf(s1[r] - mnew);
            float rs = p0 + p1;
            rs += __shfl_xor(rs, 1, 32);
            rs += __shfl_xor(rs, 2, 32);
            rs += __shfl_xor(rs, 4, 32);
            rs += __shfl_xor(rs, 8, 32);
            l_run[r] = l_run[r] * scale + rs;
#pragma unroll
            for (int t = 0; t < 16; ++t) acc[t][r] *= scale;
            // stash P tile (row-major 16x32) for the A-layout reload
            lds_p[wave][r + 8 * hi][lh]      = (__bf16)p0;
            lds_p[wave][r + 8 * hi][lh + 16] = (__bf16)p1;
        }

        asm volatile("s_wait_dscnt 0" ::: "memory");

        // ---- reload P in A layout ---------------------------------------
        v16bf pa;
        {
            const __bf16* prow = &lds_p[wave][lh][0];
            v8bf p0v = *(const v8bf*)(prow + 16 * hi);
            v8bf p1v = *(const v8bf*)(prow + 16 * hi + 8);
#pragma unroll
            for (int i = 0; i < 8; ++i) { pa[i] = p0v[i]; pa[8 + i] = p1v[i]; }
        }

        // ---- context += P * V  (K = 32 m-values) ------------------------
#pragma unroll
        for (int t = 0; t < 16; ++t) {
            const __bf16* vrow =
                xc_b + (size_t)(t * 16 + lh) * N_DIM + m0 + 16 * hi;
            v16bf vb = *(const v16bf*)vrow;
            acc[t] = wmma_bf16(pa, vb, acc[t]);
        }
    }

    // ---- finalize: divide by softmax denominator, store (B,C,N) ----------
    float linv[8];
#pragma unroll
    for (int r = 0; r < 8; ++r) linv[r] = 1.0f / l_run[r];
#pragma unroll
    for (int t = 0; t < 16; ++t) {
        const int c = t * 16 + lh;
        float* crow = ctx + (size_t)b * C_DIM * N_DIM + (size_t)c * N_DIM
                    + n_base + 8 * hi;
#pragma unroll
        for (int r = 0; r < 8; ++r) crow[r] = acc[t][r] * linv[r];
    }
}

// ---------------------------------------------------------------------------
// Kernel 3: s[b,c] = mean_n(context + x)
// ---------------------------------------------------------------------------
__global__ __launch_bounds__(256) void pool_kernel(
    const float* __restrict__ ctx, const float* __restrict__ x,
    float* __restrict__ s)
{
    __shared__ float red[256];
    const int bc = blockIdx.x;                     // 0 .. B*C-1
    const float* c1 = ctx + (size_t)bc * N_DIM;
    const float* c2 = x   + (size_t)bc * N_DIM;
    float sum = 0.0f;
    for (int i = threadIdx.x; i < N_DIM; i += 256) sum += c1[i] + c2[i];
    red[threadIdx.x] = sum;
    __syncthreads();
    for (int off = 128; off > 0; off >>= 1) {
        if (threadIdx.x < off) red[threadIdx.x] += red[threadIdx.x + off];
        __syncthreads();
    }
    if (threadIdx.x == 0) s[bc] = red[0] * (1.0f / N_DIM);
}

// ---------------------------------------------------------------------------
// Kernel 4: SE MLP: h = relu(s @ w1^T + b1); cw = sigmoid(h @ w2^T + b2)
// ---------------------------------------------------------------------------
__global__ __launch_bounds__(256) void se_kernel(
    const float* __restrict__ s,
    const float* __restrict__ w1, const float* __restrict__ b1,
    const float* __restrict__ w2, const float* __restrict__ b2,
    float* __restrict__ cw)
{
    __shared__ float ssh[C_DIM];
    __shared__ float hsh[CR_DIM];
    const int b = blockIdx.x;
    ssh[threadIdx.x] = s[b * C_DIM + threadIdx.x];
    __syncthreads();
    if (threadIdx.x < CR_DIM) {
        float a = b1[threadIdx.x];
        for (int c = 0; c < C_DIM; ++c) a += ssh[c] * w1[threadIdx.x * C_DIM + c];
        hsh[threadIdx.x] = fmaxf(a, 0.0f);
    }
    __syncthreads();
    float a = b2[threadIdx.x];
    for (int j = 0; j < CR_DIM; ++j) a += hsh[j] * w2[threadIdx.x * CR_DIM + j];
    cw[b * C_DIM + threadIdx.x] = 1.0f / (1.0f + __expf(-a));
}

// ---------------------------------------------------------------------------
// Kernel 5: out = x + gamma * context * cw[b,c]
// ---------------------------------------------------------------------------
__global__ __launch_bounds__(256) void out_kernel(
    const float* __restrict__ x, const float* __restrict__ ctx,
    const float* __restrict__ cw, const float* __restrict__ gamma,
    float* __restrict__ out)
{
    const size_t i  = (size_t)blockIdx.x * blockDim.x + threadIdx.x;
    const size_t bc = i / N_DIM;
    out[i] = x[i] + gamma[0] * ctx[i] * cw[bc];
}

// ---------------------------------------------------------------------------
extern "C" void kernel_launch(void* const* d_in, const int* in_sizes, int n_in,
                              void* d_out, int out_size, void* d_ws, size_t ws_size,
                              hipStream_t stream) {
    const float* x     = (const float*)d_in[0];
    const float* gamma = (const float*)d_in[1];
    const float* w1    = (const float*)d_in[2];
    const float* b1    = (const float*)d_in[3];
    const float* w2    = (const float*)d_in[4];
    const float* b2    = (const float*)d_in[5];
    float* out = (float*)d_out;

    // workspace carve (base is 256B-aligned; all chunks are multiples of 256B)
    char* w = (char*)d_ws;
    const size_t bnc2 = (size_t)B_DIM * N_DIM * C_DIM * 2;     // 8 MB each
    unsigned short* xT_hi = (unsigned short*)w;  w += bnc2;
    unsigned short* xT_lo = (unsigned short*)w;  w += bnc2;
    unsigned short* xc_hi = (unsigned short*)w;  w += bnc2;
    float* ctx = (float*)w;  w += (size_t)B_DIM * C_DIM * N_DIM * 4;  // 16 MB
    float* s   = (float*)w;  w += (size_t)B_DIM * C_DIM * 4;
    float* cw  = (float*)w;  w += (size_t)B_DIM * C_DIM * 4;

    ln_split_kernel<<<(B_DIM * N_DIM) / 256, 256, 0, stream>>>(
        x, xT_hi, xT_lo, xc_hi);

    attn_flash_kernel<<<(B_DIM * (N_DIM / 16)) / ATTN_WAVES, ATTN_WAVES * 32,
                        0, stream>>>(xT_hi, xT_lo, xc_hi, ctx);

    pool_kernel<<<B_DIM * C_DIM, 256, 0, stream>>>(ctx, x, s);

    se_kernel<<<B_DIM, 256, 0, stream>>>(s, w1, b1, w2, b2, cw);

    out_kernel<<<(B_DIM * C_DIM * N_DIM) / 256, 256, 0, stream>>>(
        x, ctx, cw, gamma, out);
}
